// AtlasMAGLayer_65395172049084
// MI455X (gfx1250) — compile-verified, matrix operand order checked by reading, not run on hardware
//
#include <hip/hip_runtime.h>
#include <hip/hip_bf16.h>
#include <math.h>

// ---------------------------------------------------------------------------
// Types for CDNA5 WMMA (wave32): V_WMMA_F32_16X16X32_BF16
// ---------------------------------------------------------------------------
typedef __bf16 bhalf;
typedef bhalf v16bf __attribute__((ext_vector_type(16)));
typedef float v8f   __attribute__((ext_vector_type(8)));

__device__ __forceinline__ v8f wmma_bf16(v16bf a, v16bf b, v8f c) {
  // 8 args: (neg_a, A, neg_b, B, c_mod, C, reuse_a, reuse_b)
  return __builtin_amdgcn_wmma_f32_16x16x32_bf16(false, a, false, b, (short)0, c, false, false);
}

union UA { v16bf v; bhalf e[16]; uint4 q[2]; };
union UC { v8f v; float e[8]; };

// Load one 16-element bf16 fragment whose two 8-element k-runs are contiguous
// (k-major row of stride 16B-multiple): exactly two ds_load_b128.
__device__ __forceinline__ void frag_load(UA& f, const bhalf* rowp, int khalf) {
  f.q[0] = *(const uint4*)(rowp + 8 * khalf);
  f.q[1] = *(const uint4*)(rowp + 16 + 8 * khalf);
}

// ---------------------------------------------------------------------------
// Generic mixed-precision GEMM:  C = act( alpha * op(A) @ op(B) + beta * Cin )
// op(A): MxK (transA: A stored KxM);  op(B): KxN (transB: B stored NxK)
// N % 128 == 0, K % 32 == 0, M multiple of 64 (row-predicated per 64-tile).
// 128 threads = 4 waves; block tile 64x128; each wave computes 32x64 as
// 2x4 WMMA fragments -> 8 v_wmma per k-step fed by 12 ds_load_b128, with
// 8 v8f accumulators/wave (64 VGPRs) so nothing spills to scratch.
// Both tiles staged k-contiguous (B transposed) so fragments are b128 pairs.
// ---------------------------------------------------------------------------
#define AST 40   // LDS row stride in bf16 elements (32 + 8 pad, 16B multiple)

__global__ __launch_bounds__(128) void k_gemm_bf16(
    float* __restrict__ C, const float* __restrict__ A, const float* __restrict__ Bg,
    const float* __restrict__ Cin,
    int M, int N, int K, int lda, int ldb,
    int transA, int transB, float alpha, float beta, int act)
{
  __shared__ alignas(16) bhalf As[64 * AST];    // [m][k]
  __shared__ alignas(16) bhalf Bs[128 * AST];   // [n][k]  (B staged transposed)
  const int tid = threadIdx.x;
  const int n0 = blockIdx.x * 128, m0 = blockIdx.y * 64;
  const int w = tid >> 5, lane = tid & 31;
  const int wm = (w >> 1) * 32, wn = (w & 1) * 64;
  const int khalf = lane >> 4, l15 = lane & 15;

  v8f acc[2][4] = {};

  for (int k0 = 0; k0 < K; k0 += 32) {
    if (k0 + 32 < K) {  // gfx1250 global_prefetch_b8 of next k-tiles
      __builtin_prefetch(transA ? (const void*)&A[(size_t)(k0 + 32) * lda + m0]
                                : (const void*)&A[(size_t)m0 * lda + (k0 + 32)], 0, 1);
      __builtin_prefetch(transB ? (const void*)&Bg[(size_t)n0 * ldb + (k0 + 32)]
                                : (const void*)&Bg[(size_t)(k0 + 32) * ldb + n0], 0, 1);
    }
    // ---- Stage A tile (64 m x 32 k) as [m][k], zero-fill rows >= M -----
    if (!transA) {
#pragma unroll
      for (int t = 0; t < 4; ++t) {            // 512 float4 chunks / 128 thr
        int e4 = tid + 128 * t;
        int r = e4 >> 3, kc = (e4 & 7) * 4;
        float4 f = make_float4(0.f, 0.f, 0.f, 0.f);
        if (m0 + r < M) f = *(const float4*)&A[(size_t)(m0 + r) * lda + (k0 + kc)];
        bhalf* d = &As[r * AST + kc];
        d[0] = (bhalf)f.x; d[1] = (bhalf)f.y; d[2] = (bhalf)f.z; d[3] = (bhalf)f.w;
      }
    } else {  // A stored KxM, contiguous along m: float4 along m, scatter rows
#pragma unroll
      for (int t = 0; t < 4; ++t) {
        int e4 = tid + 128 * t;
        int k = e4 >> 4, mc = (e4 & 15) * 4;   // M is a multiple of 64
        float4 f = make_float4(0.f, 0.f, 0.f, 0.f);
        if (m0 + mc < M) f = *(const float4*)&A[(size_t)(k0 + k) * lda + (m0 + mc)];
        As[(mc + 0) * AST + k] = (bhalf)f.x;
        As[(mc + 1) * AST + k] = (bhalf)f.y;
        As[(mc + 2) * AST + k] = (bhalf)f.z;
        As[(mc + 3) * AST + k] = (bhalf)f.w;
      }
    }
    // ---- Stage B tile (32 k x 128 n) as [n][k] -------------------------
    if (transB) {  // B stored NxK: contiguous along k
#pragma unroll
      for (int t = 0; t < 8; ++t) {            // 1024 float4 chunks / 128 thr
        int e4 = tid + 128 * t;
        int n = e4 >> 3, kc = (e4 & 7) * 4;
        float4 f = *(const float4*)&Bg[(size_t)(n0 + n) * ldb + (k0 + kc)];
        bhalf* d = &Bs[n * AST + kc];
        d[0] = (bhalf)f.x; d[1] = (bhalf)f.y; d[2] = (bhalf)f.z; d[3] = (bhalf)f.w;
      }
    } else {  // B stored KxN: contiguous along n, scatter to [n][k]
#pragma unroll
      for (int t = 0; t < 8; ++t) {
        int e4 = tid + 128 * t;
        int k = e4 >> 5, nc = (e4 & 31) * 4;
        float4 f = *(const float4*)&Bg[(size_t)(k0 + k) * ldb + (n0 + nc)];
        Bs[(nc + 0) * AST + k] = (bhalf)f.x;
        Bs[(nc + 1) * AST + k] = (bhalf)f.y;
        Bs[(nc + 2) * AST + k] = (bhalf)f.z;
        Bs[(nc + 3) * AST + k] = (bhalf)f.w;
      }
    }
    __syncthreads();

    // 2 A-fragments held in registers; each B-fragment reused across 2 rows.
    UA afr[2];
#pragma unroll
    for (int i2 = 0; i2 < 2; ++i2)
      frag_load(afr[i2], &As[(wm + i2 * 16 + l15) * AST], khalf);
#pragma unroll
    for (int j2 = 0; j2 < 4; ++j2) {
      UA b;
      frag_load(b, &Bs[(wn + j2 * 16 + l15) * AST], khalf);
#pragma unroll
      for (int i2 = 0; i2 < 2; ++i2)
        acc[i2][j2] = wmma_bf16(afr[i2].v, b.v, acc[i2][j2]);
    }
    __syncthreads();
  }

  // Epilogue: C/D layout -> vgpr r: lanes0-15 M=r, lanes16-31 M=r+8.
  for (int i2 = 0; i2 < 2; ++i2)
    for (int j2 = 0; j2 < 4; ++j2) {
      UC u; u.v = acc[i2][j2];
      const int col = n0 + wn + j2 * 16 + l15;
#pragma unroll
      for (int r = 0; r < 8; ++r) {
        const int row = m0 + wm + i2 * 16 + r + 8 * khalf;
        if (row < M) {
          float vv = alpha * u.e[r];
          if (beta != 0.0f) vv += beta * Cin[(size_t)row * N + col];
          if (act) vv = vv / (1.0f + __expf(-vv));  // silu
          C[(size_t)row * N + col] = vv;
        }
      }
    }
}

// ---------------------------------------------------------------------------
// Sliding-window attention core (one block per (b, h, 64-query block)).
// Fixed 192-key window [q0-128, q0+64) staged in LDS (bf16); QK^T and P*V via
// WMMA bf16; f32 softmax with causal + 128-window mask. D=512, H=8, hd=64.
// K staged [key][k]; V staged transposed [c][key]; P [row][key] -> all
// fragments are pairs of ds_load_b128.
// ---------------------------------------------------------------------------
#define KST 72    // K rows: 64 + 8 pad (bf16, 144B = 16B multiple)
#define SST 200   // score/prob/Vt rows: 192 + 8 pad (400B f32-stride also ok)
static const int SWA_LDS_BYTES = 192 * KST * 2    // Kb
                               + 64 * SST * 2     // Vt
                               + 64 * SST * 4     // Sf
                               + 64 * SST * 2;    // Pb

__global__ __launch_bounds__(128) void k_swa(
    const float* __restrict__ Q, const float* __restrict__ Kg,
    const float* __restrict__ V, float* __restrict__ O, int S, int nqb)
{
  extern __shared__ char smem[];
  bhalf* Kb = (bhalf*)smem;                  // [192][KST]
  bhalf* Vt = Kb + 192 * KST;                // [64][SST]  (V transposed)
  float* Sf = (float*)(Vt + 64 * SST);       // [64][SST]
  bhalf* Pb = (bhalf*)(Sf + 64 * SST);       // [64][SST]

  const int tid = threadIdx.x;
  const int qb = blockIdx.x % nqb;
  const int h  = (blockIdx.x / nqb) % 8;
  const int b  = blockIdx.x / (nqb * 8);
  const int q0 = qb * 64;
  const size_t baseBH = (size_t)b * S * 512 + (size_t)h * 64;

  // Load 192x64 K and V windows with float4 (zero-pad j<0); V transposed.
  for (int t = 0; t < 24; ++t) {             // 3072 float4 chunks / 128 thr
    int e4 = tid + 128 * t;
    int j = e4 >> 4, c4 = (e4 & 15) * 4;
    int jabs = q0 - 128 + j;
    float4 fk = make_float4(0.f, 0.f, 0.f, 0.f);
    float4 fv = make_float4(0.f, 0.f, 0.f, 0.f);
    if (jabs >= 0) {
      fk = *(const float4*)&Kg[baseBH + (size_t)jabs * 512 + c4];
      fv = *(const float4*)&V [baseBH + (size_t)jabs * 512 + c4];
    }
    bhalf* kd = &Kb[j * KST + c4];
    kd[0] = (bhalf)fk.x; kd[1] = (bhalf)fk.y; kd[2] = (bhalf)fk.z; kd[3] = (bhalf)fk.w;
    Vt[(c4 + 0) * SST + j] = (bhalf)fv.x;
    Vt[(c4 + 1) * SST + j] = (bhalf)fv.y;
    Vt[(c4 + 2) * SST + j] = (bhalf)fv.z;
    Vt[(c4 + 3) * SST + j] = (bhalf)fv.w;
  }
  __syncthreads();

  const int w = tid >> 5, lane = tid & 31, khalf = lane >> 4, l15 = lane & 15;
  const int wq0 = w * 16;

  // Q fragments from global via float4 (16 rows x 64 k, 2 k-steps).
  UA aq[2];
  {
    const float* qrow = Q + baseBH + (size_t)(q0 + wq0 + l15) * 512;
#pragma unroll
    for (int ks = 0; ks < 2; ++ks)
#pragma unroll
      for (int hseg = 0; hseg < 2; ++hseg) {
        const float* p = qrow + 32 * ks + hseg * 16 + 8 * khalf;
        float4 f0 = *(const float4*)p;
        float4 f1 = *(const float4*)(p + 4);
        bhalf* d = &aq[ks].e[hseg * 8];
        d[0] = (bhalf)f0.x; d[1] = (bhalf)f0.y; d[2] = (bhalf)f0.z; d[3] = (bhalf)f0.w;
        d[4] = (bhalf)f1.x; d[5] = (bhalf)f1.y; d[6] = (bhalf)f1.z; d[7] = (bhalf)f1.w;
      }
  }

  // Scores: 16x192 per wave = 12 tiles x 2 k-steps of WMMA.
  for (int n = 0; n < 12; ++n) {
    v8f acc = {};
#pragma unroll
    for (int ks = 0; ks < 2; ++ks) {
      UA bk;
      frag_load(bk, &Kb[(n * 16 + l15) * KST + 32 * ks], khalf);
      acc = wmma_bf16(aq[ks].v, bk.v, acc);
    }
    UC u; u.v = acc;
#pragma unroll
    for (int r = 0; r < 8; ++r)
      Sf[(wq0 + r + 8 * khalf) * SST + n * 16 + l15] = u.e[r];
  }
  __syncthreads();

  // Softmax (threads 0..63 each own one query row).
  if (tid < 64) {
    const int iabs = q0 + tid;
    float mx = -1e30f;
    for (int j = 0; j < 192; ++j) {
      int jabs = q0 - 128 + j;
      bool ok = (jabs >= 0) && (jabs <= iabs) && (iabs - jabs < 128);
      float s = ok ? Sf[tid * SST + j] * 0.125f : -1e30f;
      Sf[tid * SST + j] = s;
      mx = fmaxf(mx, s);
    }
    float sum = 0.f;
    for (int j = 0; j < 192; ++j) {
      float p = __expf(Sf[tid * SST + j] - mx);
      sum += p;
      Sf[tid * SST + j] = p;
    }
    float inv = 1.0f / sum;
    for (int j = 0; j < 192; ++j)
      Pb[tid * SST + j] = (bhalf)(Sf[tid * SST + j] * inv);
  }
  __syncthreads();

  // O = P(16x192) @ V(192x64): 4 tiles x 6 k-steps per wave.
  for (int n = 0; n < 4; ++n) {
    v8f acc = {};
#pragma unroll
    for (int ks = 0; ks < 6; ++ks) {
      UA ap, bv;
      frag_load(ap, &Pb[(wq0 + l15) * SST + 32 * ks], khalf);
      frag_load(bv, &Vt[(n * 16 + l15) * SST + 32 * ks], khalf);
      acc = wmma_bf16(ap.v, bv.v, acc);
    }
    UC u; u.v = acc;
#pragma unroll
    for (int r = 0; r < 8; ++r) {
      int row = q0 + wq0 + r + 8 * khalf;
      O[baseBH + (size_t)row * 512 + n * 16 + l15] = u.e[r];
    }
  }
}

// ---------------------------------------------------------------------------
// Small elementwise / reduction kernels
// ---------------------------------------------------------------------------
__global__ __launch_bounds__(256) void k_rms(const float* __restrict__ in,
                                             const float* __restrict__ wt,
                                             float* __restrict__ out, int Dd) {
  __shared__ float red[256];
  const int row = blockIdx.x;
  const float* x = in + (size_t)row * Dd;
  float s = 0.f;
  for (int d = threadIdx.x; d < Dd; d += 256) s += x[d] * x[d];
  red[threadIdx.x] = s; __syncthreads();
  for (int o = 128; o > 0; o >>= 1) {
    if (threadIdx.x < o) red[threadIdx.x] += red[threadIdx.x + o];
    __syncthreads();
  }
  const float inv = rsqrtf(red[0] / (float)Dd + 1e-6f);
  for (int d = threadIdx.x; d < Dd; d += 256)
    out[(size_t)row * Dd + d] = x[d] * wt[d] * inv;
}

__global__ void k_dwconv_silu(const float* __restrict__ in, const float* __restrict__ w3,
                              const float* __restrict__ bias, float* __restrict__ out,
                              int Bn, int S, int Dd) {
  int idx = blockIdx.x * blockDim.x + threadIdx.x;
  if (idx >= Bn * S * Dd) return;
  int d = idx % Dd, s = (idx / Dd) % S, b = idx / (Dd * S);
  const float* base = in + (size_t)b * S * Dd;
  float xm = (s > 0)     ? base[(size_t)(s - 1) * Dd + d] : 0.f;
  float xc =               base[(size_t)s * Dd + d];
  float xp = (s < S - 1) ? base[(size_t)(s + 1) * Dd + d] : 0.f;
  float v = xm * w3[d * 3 + 0] + xc * w3[d * 3 + 1] + xp * w3[d * 3 + 2] + bias[d];
  out[idx] = v / (1.f + __expf(-v));
}

__global__ void k_zero(float* p, int n) {
  int i = blockIdx.x * blockDim.x + threadIdx.x;
  if (i < n) p[i] = 0.f;
}

__global__ __launch_bounds__(256) void k_sqsum(const float* __restrict__ x, int n,
                                               float* __restrict__ out) {
  __shared__ float red[256];
  float s = 0.f;
  for (int i = blockIdx.x * 256 + threadIdx.x; i < n; i += gridDim.x * 256) s += x[i] * x[i];
  red[threadIdx.x] = s; __syncthreads();
  for (int o = 128; o > 0; o >>= 1) {
    if (threadIdx.x < o) red[threadIdx.x] += red[threadIdx.x + o];
    __syncthreads();
  }
  if (threadIdx.x == 0) atomicAdd(out, red[0]);
}

__global__ void k_scale_bynorm(const float* __restrict__ g, float* __restrict__ xo,
                               const float* __restrict__ ssq, int n) {
  int i = blockIdx.x * blockDim.x + threadIdx.x;
  if (i >= n) return;
  float inv = 1.0f / (sqrtf(*ssq) + 1e-7f);
  xo[i] = g[i] * inv;
}

__global__ void k_colmean(const float* __restrict__ g, float* __restrict__ meanE,
                          float* __restrict__ meanA, int ld, int Dd) {
  int d = blockIdx.x * blockDim.x + threadIdx.x;
  if (d >= Dd) return;
  float se = 0.f, sa = 0.f;
  for (int r = 0; r < 64; ++r) {
    se += g[(size_t)r * ld + Dd + d];       // eta
    sa += g[(size_t)r * ld + 2 * Dd + d];   // alpha
  }
  meanE[d] = se * (1.0f / 64.0f);
  meanA[d] = sa * (1.0f / 64.0f);
}

__global__ void k_mupdate(float* __restrict__ M, const float* __restrict__ og,
                          const float* __restrict__ mE, const float* __restrict__ mA,
                          int n, int Dd) {
  int i = blockIdx.x * blockDim.x + threadIdx.x;
  if (i >= n) return;
  int e = i % Dd;
  M[i] = M[i] * mA[e] - og[i] * mE[e];
}

__global__ void k_gatemul(const float* __restrict__ a, const float* __restrict__ bp,
                          const float* __restrict__ g, float* __restrict__ o,
                          int Dd, int n) {
  int i = blockIdx.x * blockDim.x + threadIdx.x;
  if (i >= n) return;
  int row = i / Dd, d = i % Dd;
  o[i] = a[i] * bp[i] * g[(size_t)row * 3 * Dd + d];  // gamma = gates[:, :D]
}

__global__ void k_add(const float* __restrict__ a, const float* __restrict__ b,
                      float* __restrict__ o, int n) {
  int i = blockIdx.x * blockDim.x + threadIdx.x;
  if (i < n) o[i] = a[i] + b[i];
}

__global__ void k_mul(const float* __restrict__ a, const float* __restrict__ b,
                      float* __restrict__ o, int n) {
  int i = blockIdx.x * blockDim.x + threadIdx.x;
  if (i < n) o[i] = a[i] * b[i];
}

// ---------------------------------------------------------------------------
// Host orchestration
// ---------------------------------------------------------------------------
extern "C" void kernel_launch(void* const* d_in, const int* in_sizes, int n_in,
                              void* d_out, int out_size, void* d_ws, size_t ws_size,
                              hipStream_t stream) {
  (void)in_sizes; (void)n_in; (void)out_size; (void)ws_size;
  const int Bn = 2, S = 2048, D = 512, H = 8, CH = 64, WIN = 128, FF = 2048, NC = 32;
  const size_t BS = (size_t)Bn * S, BSD = BS * D, D2 = (size_t)D * D;

  const float* x         = (const float*)d_in[0];
  const float* mem_state = (const float*)d_in[1];
  const float* buf_k     = (const float*)d_in[2];
  const float* buf_v     = (const float*)d_in[3];
  const float* nw_in     = (const float*)d_in[4];
  const float* nw_kq     = (const float*)d_in[5];
  const float* nw_out    = (const float*)d_in[6];
  const float* wk_a      = (const float*)d_in[7];
  const float* wq_a      = (const float*)d_in[8];
  const float* wv_a      = (const float*)d_in[9];
  const float* wg        = (const float*)d_in[10];
  const float* wb        = (const float*)d_in[11];
  const float* ck_w      = (const float*)d_in[12];
  const float* ck_b      = (const float*)d_in[13];
  const float* cq_w      = (const float*)d_in[14];
  const float* cq_b      = (const float*)d_in[15];
  const float* s1_wq     = (const float*)d_in[16];
  const float* s1_wk     = (const float*)d_in[17];
  const float* s1_wv     = (const float*)d_in[18];
  const float* s1_wo     = (const float*)d_in[19];
  const float* s2_wq     = (const float*)d_in[20];
  const float* s2_wk     = (const float*)d_in[21];
  const float* s2_wv     = (const float*)d_in[22];
  const float* s2_wo     = (const float*)d_in[23];
  const float* n1_w      = (const float*)d_in[24];
  const float* n2_w      = (const float*)d_in[25];
  const float* m_w1      = (const float*)d_in[26];
  const float* m_w2      = (const float*)d_in[27];
  const float* m_w3      = (const float*)d_in[28];

  float* out   = (float*)d_out;
  float* outM  = out + BSD;                 // Mf [B,D,D] — scan state lives here
  float* outBK = outM + (size_t)Bn * D2;    // bkf [B,WIN,D]
  float* outBV = outBK + (size_t)Bn * WIN * D;

  float* w = (float*)d_ws; size_t off = 0;
  auto F = [&](size_t n) { float* p = w + off; off += n; return p; };
  float* xn    = F(BSD);
  float* vb    = F(BSD);
  float* kb    = F(BSD);
  float* qb    = F(BSD);
  float* tmp   = F(BSD);
  float* memo  = F(BSD);
  float* gates = F(3 * BSD);
  float* byp   = F(BSD);
  float* araw  = F(BSD);
  float* aq    = F(BSD);
  float* ak    = F(BSD);
  float* av    = F(BSD);
  float* ao    = F(BSD);
  float* attn1 = F(BSD);
  float* fused = F(BSD);
  float* yb    = F(BSD);
  float* h1    = F(BS * FF);
  float* h2    = F(BS * FF);
  float* ctxk  = F((size_t)192 * D);
  float* ctxv  = F((size_t)192 * D);
  float* errb  = F((size_t)192 * D);
  float* Xa    = F(D2);
  float* Xb    = F(D2);
  float* Am    = F(D2);
  float* Bmb   = F(D2);
  float* meanE = F(D);
  float* meanA = F(D);
  float* ssq   = F(1);

  auto gemm = [&](float* C, const float* A, const float* Bp, int M, int N, int K,
                  int lda, int ldb, int tA, int tB, float al, float be,
                  const float* Cin, int act) {
    k_gemm_bf16<<<dim3(N / 128, (M + 63) / 64), dim3(128), 0, stream>>>(
        C, A, Bp, Cin, M, N, K, lda, ldb, tA, tB, al, be, act);
  };
  const int EW = 256;
  auto nb = [](size_t n) { return (unsigned)((n + 255) / 256); };

  // ---- Atlas front end -----------------------------------------------------
  k_rms<<<(unsigned)BS, EW, 0, stream>>>(x, nw_in, xn, D);
  gemm(vb, xn, wv_a, (int)BS, D, D, D, D, 0, 1, 1.f, 0.f, nullptr, 1);        // v = silu(xn Wv^T)
  gemm(tmp, xn, wk_a, (int)BS, D, D, D, D, 0, 1, 1.f, 0.f, nullptr, 0);
  k_dwconv_silu<<<nb(BSD), EW, 0, stream>>>(tmp, ck_w, ck_b, memo, Bn, S, D);
  k_rms<<<(unsigned)BS, EW, 0, stream>>>(memo, nw_kq, kb, D);                 // k
  gemm(tmp, xn, wq_a, (int)BS, D, D, D, D, 0, 1, 1.f, 0.f, nullptr, 0);
  k_dwconv_silu<<<nb(BSD), EW, 0, stream>>>(tmp, cq_w, cq_b, memo, Bn, S, D);
  k_rms<<<(unsigned)BS, EW, 0, stream>>>(memo, nw_kq, qb, D);                 // q
  gemm(gates, xn, wg, (int)BS, 3 * D, D, D, D, 0, 1, 1.f, 0.f, nullptr, 1);   // silu gates
  gemm(byp, xn, wb, (int)BS, D, D, D, D, 0, 1, 1.f, 0.f, nullptr, 1);         // bypass

  // ---- Chunked scan with Newton–Schulz (bf16 WMMA throughout) -------------
  hipMemcpyAsync(outM, mem_state, (size_t)Bn * D2 * 4, hipMemcpyDeviceToDevice, stream);
  hipMemcpyAsync(outBK, buf_k, (size_t)Bn * WIN * D * 4, hipMemcpyDeviceToDevice, stream);
  hipMemcpyAsync(outBV, buf_v, (size_t)Bn * WIN * D * 4, hipMemcpyDeviceToDevice, stream);

  const float nsA = 3.4445f, nsB = -4.7750f, nsC = 2.0315f;
  for (int b = 0; b < Bn; ++b) {
    float* Mb  = outM + (size_t)b * D2;
    float* bkb = outBK + (size_t)b * WIN * D;
    float* bvb = outBV + (size_t)b * WIN * D;
    for (int c = 0; c < NC; ++c) {
      const size_t co = ((size_t)b * S + (size_t)c * CH) * D;
      hipMemcpyAsync(ctxk, bkb, (size_t)WIN * D * 4, hipMemcpyDeviceToDevice, stream);
      hipMemcpyAsync(ctxk + (size_t)WIN * D, kb + co, (size_t)CH * D * 4, hipMemcpyDeviceToDevice, stream);
      hipMemcpyAsync(ctxv, bvb, (size_t)WIN * D * 4, hipMemcpyDeviceToDevice, stream);
      hipMemcpyAsync(ctxv + (size_t)WIN * D, vb + co, (size_t)CH * D * 4, hipMemcpyDeviceToDevice, stream);

      gemm(errb, ctxk, Mb, 192, D, D, D, D, 0, 0, 1.f, -1.f, ctxv, 0);        // err = K·M − V
      gemm(Am, ctxk, errb, D, D, 192, D, D, 1, 0, 1.f, 0.f, nullptr, 0);      // grad = K^T·err
      k_zero<<<1, 1, 0, stream>>>(ssq, 1);
      k_sqsum<<<256, 256, 0, stream>>>(Am, (int)D2, ssq);
      k_scale_bynorm<<<nb(D2), EW, 0, stream>>>(Am, Xa, ssq, (int)D2);        // X = G/(|G|+1e-7)

      float* Xc = Xa; float* Xn2 = Xb;
      for (int it = 0; it < 5; ++it) {
        gemm(Am, Xc, Xc, D, D, D, D, D, 0, 1, 1.f, 0.f, nullptr, 0);          // A = X·X^T
        gemm(Bmb, Am, Am, D, D, D, D, D, 0, 0, nsC, nsB, Am, 0);              // Bm = c·A·A + b·A
        gemm(Xn2, Bmb, Xc, D, D, D, D, D, 0, 0, 1.f, nsA, Xc, 0);             // X = Bm·X + a·X
        float* t2 = Xc; Xc = Xn2; Xn2 = t2;
      }
      k_colmean<<<2, 256, 0, stream>>>(gates + ((size_t)b * S + (size_t)c * CH) * 3 * D,
                                       meanE, meanA, 3 * D, D);
      k_mupdate<<<nb(D2), EW, 0, stream>>>(Mb, Xc, meanE, meanA, (int)D2, D); // M = M·ᾱ − og·η̄
      gemm(araw + co, qb + co, Mb, CH, D, D, D, D, 0, 0, 1.f, 0.f, nullptr, 0);

      hipMemcpyAsync(bkb, ctxk + (size_t)CH * D, (size_t)WIN * D * 4, hipMemcpyDeviceToDevice, stream);
      hipMemcpyAsync(bvb, ctxv + (size_t)CH * D, (size_t)WIN * D * 4, hipMemcpyDeviceToDevice, stream);
    }
  }

  // ---- Atlas epilogue: rms * bypass * gamma, then n1 rms -------------------
  k_rms<<<(unsigned)BS, EW, 0, stream>>>(araw, nw_out, tmp, D);
  k_gatemul<<<nb(BSD), EW, 0, stream>>>(tmp, byp, gates, araw, D, (int)BSD);
  k_rms<<<(unsigned)BS, EW, 0, stream>>>(araw, n1_w, memo, D);                // mem_out

  // ---- Sliding-window attention helper ------------------------------------
  auto swa = [&](const float* inX, const float* wq_, const float* wk_,
                 const float* wv_, const float* wo_, float* outp) {
    gemm(aq, inX, wq_, (int)BS, D, D, D, D, 0, 1, 1.f, 0.f, nullptr, 0);
    gemm(ak, inX, wk_, (int)BS, D, D, D, D, 0, 1, 1.f, 0.f, nullptr, 0);
    gemm(av, inX, wv_, (int)BS, D, D, D, D, 0, 1, 1.f, 0.f, nullptr, 0);
    k_swa<<<Bn * H * (S / 64), 128, SWA_LDS_BYTES, stream>>>(aq, ak, av, ao, S, S / 64);
    gemm(outp, ao, wo_, (int)BS, D, D, D, D, 0, 1, 1.f, 0.f, nullptr, 0);
  };

  swa(x, s1_wq, s1_wk, s1_wv, s1_wo, tmp);
  k_rms<<<(unsigned)BS, EW, 0, stream>>>(tmp, n2_w, attn1, D);
  k_add<<<nb(BSD), EW, 0, stream>>>(memo, attn1, fused, (int)BSD);            // fused

  swa(fused, s2_wq, s2_wk, s2_wv, s2_wo, yb);                                  // y

  // ---- SwiGLU MLP ----------------------------------------------------------
  gemm(h1, yb, m_w1, (int)BS, FF, D, D, D, 0, 1, 1.f, 0.f, nullptr, 1);       // silu(y W1^T)
  gemm(h2, yb, m_w2, (int)BS, FF, D, D, D, 0, 1, 1.f, 0.f, nullptr, 0);       // y W2^T
  k_mul<<<nb(BS * FF), EW, 0, stream>>>(h1, h2, h1, (int)(BS * FF));
  gemm(tmp, h1, m_w3, (int)BS, D, FF, FF, FF, 0, 1, 1.f, 0.f, nullptr, 0);    // @ W3^T
  k_add<<<nb(BSD), EW, 0, stream>>>(fused, tmp, out, (int)BSD);               // out
}
